// GRUFusion_36378372997204
// MI455X (gfx1250) — matrix operand order repaired and checked by reading, not compiled, not run on hardware
//
#include <hip/hip_runtime.h>
#include <hip/hip_bf16.h>

#define DIM 96
#define CCH 24
#define TOTAL (DIM * DIM * DIM)          /* 884736 */
#define NTILES (TOTAL / 16)              /* 55296  */
#define WAVES_PER_BLOCK 4
#define GRU_BLOCKS (NTILES / WAVES_PER_BLOCK) /* 13824 */
#define MASK_BLOCK 256
#define NMASKBLK (TOTAL / MASK_BLOCK)    /* 3456 */
#define VOXEL_SIZE 0.04f

typedef __attribute__((ext_vector_type(16))) _Float16 v16h;
typedef __attribute__((ext_vector_type(8)))  _Float16 v8h;
typedef __attribute__((ext_vector_type(8)))  float    v8f;

// ---------------------------------------------------------------- utilities
__device__ __forceinline__ float fast_rcp(float x) {
    return __builtin_amdgcn_rcpf(x);
}
__device__ __forceinline__ float sigmoidf_(float x) {
    return fast_rcp(1.0f + __expf(-x));
}
__device__ __forceinline__ float tanhf_(float x) {
    return 1.0f - 2.0f * fast_rcp(1.0f + __expf(2.0f * x));
}

__device__ __forceinline__ v8f wmma_f16(v16h a, v16h b, v8f c) {
    // D = A(16x32 f16) * B(32x16 f16) + C(16x16 f32)
    return __builtin_amdgcn_wmma_f32_16x16x32_f16(
        /*neg_a=*/false, a, /*neg_b=*/false, b,
        /*c_mod=*/(short)0, c, /*reuse_a=*/false, /*reuse_b=*/false);
}

// Async global->LDS copy, 16B per lane; INST_OFFSET applies to both sides.
__device__ __forceinline__ void async_b128(unsigned int lds_addr, unsigned long long gaddr) {
    asm volatile("global_load_async_to_lds_b128 %0, %1, off"
                 :: "v"(lds_addr), "v"(gaddr) : "memory");
}
__device__ __forceinline__ void async_b128_off16(unsigned int lds_addr, unsigned long long gaddr) {
    asm volatile("global_load_async_to_lds_b128 %0, %1, off offset:16"
                 :: "v"(lds_addr), "v"(gaddr) : "memory");
}
__device__ __forceinline__ void async_b128_off32(unsigned int lds_addr, unsigned long long gaddr) {
    asm volatile("global_load_async_to_lds_b128 %0, %1, off offset:32"
                 :: "v"(lds_addr), "v"(gaddr) : "memory");
}
__device__ __forceinline__ void wait_async0() {
    asm volatile("s_wait_asynccnt 0x0" ::: "memory");
}
__device__ __forceinline__ unsigned int lds_addr_of(const void* p) {
    // generic "shared" addresses carry the LDS byte offset in the low 32 bits
    return (unsigned int)(unsigned long long)(uintptr_t)p;
}

// A-matrix fragment (16x32 f16) from a row-major 16x64 LDS tile.
// Layout: lanes 0-15 row M=lane, v[0..15]=K 0..15;
// lanes 16-31 row M=lane-16, v[0..15]=K 16..31.
__device__ __forceinline__ v16h load_a_frag(const _Float16* tile, int lane, int kOffset) {
    int m  = lane & 15;
    int b0 = (lane >> 4) * 16;
    const v8h* p0 = (const v8h*)(tile + m * 64 + kOffset + b0);
    const v8h* p1 = (const v8h*)(tile + m * 64 + kOffset + 8 + b0);
    v8h lo = *p0;
    v8h hi = *p1;
    v16h a;
#pragma unroll
    for (int i = 0; i < 8; ++i) { a[i] = lo[i]; a[8 + i] = hi[i]; }
    return a;
}

// B-matrix fragment (32x16 f16) from a column-major [col][k] LDS tile (64 K slots).
// Layout: lane L -> column N = L&15; lanes 0-15 hold K 0..15, lanes 16-31 K 16..31.
__device__ __forceinline__ v16h load_b_frag(const _Float16* Bt, int lane, int ntile, int kOffset) {
    int col = ntile * 16 + (lane & 15);
    int k0  = kOffset + 16 * (lane >> 4);
    return *(const v16h*)(Bt + col * 64 + k0);
}

// ---------------------------------------------------------------- kernels
__global__ void k_zero(float4* vols, int nv4, int4* ulist, int nu4) {
    int i = blockIdx.x * blockDim.x + threadIdx.x;
    int stride = gridDim.x * blockDim.x;
    float4 z4 = {0.f, 0.f, 0.f, 0.f};
    int4 zi4 = {0, 0, 0, 0};
    for (int t = i; t < nv4; t += stride) vols[t] = z4;
    for (int t = i; t < nu4; t += stride) ulist[t] = zi4;
}

__global__ void k_scatter_current(const int* __restrict__ coords,
                                  const float* __restrict__ vals,
                                  float* __restrict__ cvol, int N) {
    int i = blockIdx.x * blockDim.x + threadIdx.x;
    if (i >= N) return;
    int x = coords[i * 4 + 1];
    int y = coords[i * 4 + 2];
    int z = coords[i * 4 + 3];
    if ((unsigned)x >= DIM || (unsigned)y >= DIM || (unsigned)z >= DIM) return;
    float4* d = (float4*)(cvol + (size_t)((x * DIM + y) * DIM + z) * CCH);
    const float4* s = (const float4*)(vals + (size_t)i * CCH);
#pragma unroll
    for (int j = 0; j < 6; ++j) d[j] = s[j];
}

__global__ void k_scatter_global(const int* __restrict__ gcoords,
                                 const float* __restrict__ gvals,
                                 const int* __restrict__ relorigin,
                                 float* __restrict__ gvol, int M) {
    int i = blockIdx.x * blockDim.x + threadIdx.x;
    if (i >= M) return;
    int x = gcoords[i * 3 + 0] - relorigin[0];
    int y = gcoords[i * 3 + 1] - relorigin[1];
    int z = gcoords[i * 3 + 2] - relorigin[2];
    if ((unsigned)x >= DIM || (unsigned)y >= DIM || (unsigned)z >= DIM) return;
    float4* d = (float4*)(gvol + (size_t)((x * DIM + y) * DIM + z) * CCH);
    const float4* s = (const float4*)(gvals + (size_t)i * CCH);
#pragma unroll
    for (int j = 0; j < 6; ++j) d[j] = s[j];
}

__global__ void k_mask(const float* __restrict__ gvol, const float* __restrict__ cvol,
                       unsigned char* __restrict__ mask, int* __restrict__ bcnt) {
    int v = blockIdx.x * MASK_BLOCK + threadIdx.x;
    const float4* g = (const float4*)(gvol + (size_t)v * CCH);
    const float4* c = (const float4*)(cvol + (size_t)v * CCH);
    int m = 0;
#pragma unroll
    for (int j = 0; j < 6; ++j) {
        float4 a = g[j], b = c[j];
        m |= (a.x != 0.f) | (a.y != 0.f) | (a.z != 0.f) | (a.w != 0.f) |
             (b.x != 0.f) | (b.y != 0.f) | (b.z != 0.f) | (b.w != 0.f);
    }
    mask[v] = (unsigned char)m;
    __shared__ int cnt;
    if (threadIdx.x == 0) cnt = 0;
    __syncthreads();
    if (m) atomicAdd(&cnt, 1);
    __syncthreads();
    if (threadIdx.x == 0) bcnt[blockIdx.x] = cnt;
}

__global__ void k_scan(const int* __restrict__ bcnt, int* __restrict__ boff,
                       int* __restrict__ totalOut) {
    const int NB = NMASKBLK;
    int t = threadIdx.x;
    int c[4];
    int partial = 0;
#pragma unroll
    for (int j = 0; j < 4; ++j) {
        int idx = t * 4 + j;
        c[j] = (idx < NB) ? bcnt[idx] : 0;
        partial += c[j];
    }
    __shared__ int s[1024];
    s[t] = partial;
    __syncthreads();
    for (int off = 1; off < 1024; off <<= 1) {
        int v = (t >= off) ? s[t - off] : 0;
        __syncthreads();
        s[t] += v;
        __syncthreads();
    }
    int base = s[t] - partial;
#pragma unroll
    for (int j = 0; j < 4; ++j) {
        int idx = t * 4 + j;
        if (idx < NB) { boff[idx] = base; base += c[j]; }
    }
    if (t == 1023) totalOut[0] = s[1023];
}

__global__ void k_compact(const unsigned char* __restrict__ mask,
                          const int* __restrict__ boff, int* __restrict__ ulist) {
    int v = blockIdx.x * MASK_BLOCK + threadIdx.x;
    int m = mask[v];
    unsigned long long bal = __ballot(m != 0);
    int lane = threadIdx.x & 31;
    int wid  = threadIdx.x >> 5;
    int prefix = __popcll(bal & ((1ull << lane) - 1ull));
    __shared__ int wtot[8], woff[8];
    if (lane == 0) wtot[wid] = __popcll(bal);
    __syncthreads();
    if (threadIdx.x == 0) {
        int r = 0;
        for (int i = 0; i < 8; ++i) { woff[i] = r; r += wtot[i]; }
    }
    __syncthreads();
    if (m) ulist[boff[blockIdx.x] + woff[wid] + prefix] = v;
}

__global__ void k_coords(const int* __restrict__ ulist, const int* __restrict__ totalPtr,
                         const float* __restrict__ origin, const float* __restrict__ w2ac,
                         float* __restrict__ out) {
    int t = blockIdx.x * 256 + threadIdx.x;
    if (t >= TOTAL) return;
    int v = ulist[t];
    int ux = v / (DIM * DIM);
    int rem = v - ux * DIM * DIM;
    int uy = rem / DIM;
    int uz = rem - uy * DIM;

    float* oc  = out;                         // updated_coords_all (TOTAL,4)
    float* orc = out + (size_t)TOTAL * 4;     // r_coords (TOTAL,4)
    float* ov  = out + (size_t)TOTAL * 32;    // row_valid (TOTAL,)

    oc[(size_t)t * 4 + 0] = 0.0f;
    oc[(size_t)t * 4 + 1] = (float)ux;
    oc[(size_t)t * 4 + 2] = (float)uy;
    oc[(size_t)t * 4 + 3] = (float)uz;

    float wx = (float)ux * VOXEL_SIZE + origin[0];
    float wy = (float)uy * VOXEL_SIZE + origin[1];
    float wz = (float)uz * VOXEL_SIZE + origin[2];
#pragma unroll
    for (int i = 0; i < 3; ++i) {
        float cam = w2ac[i * 4 + 0] * wx + w2ac[i * 4 + 1] * wy +
                    w2ac[i * 4 + 2] * wz + w2ac[i * 4 + 3];
        orc[(size_t)t * 4 + i] = cam;
    }
    orc[(size_t)t * 4 + 3] = 0.0f;
    ov[t] = (t < totalPtr[0]) ? 1.0f : 0.0f;
}

__global__ __launch_bounds__(128)
void k_gru(const float* __restrict__ gvol, const float* __restrict__ cvol,
           const int* __restrict__ ulist, const int* __restrict__ totalPtr,
           const float* __restrict__ Wz, const float* __restrict__ bz,
           const float* __restrict__ Wr, const float* __restrict__ br,
           const float* __restrict__ Wq, const float* __restrict__ bq,
           float* __restrict__ outFused) {
    __shared__ _Float16 Bw[3][32 * 64];                 // weights, col-major [col][k], 12 KB
    __shared__ _Float16 hx[WAVES_PER_BLOCK][16 * 64];   // A tile [h | x | 0-pad], 8 KB
    __shared__ float    stg[WAVES_PER_BLOCK][16 * 48];  // async-staged f32 [h(24)|x(24)], 12 KB

    const int tid = threadIdx.x, lane = tid & 31, wave = tid >> 5;
    const int tileBase = (blockIdx.x * WAVES_PER_BLOCK + wave) * 16;
    __builtin_prefetch(ulist + tileBase + 64, 0, 1);    // -> global_prefetch_b8

    // Kick off the gather first: 2 lanes per row, 12 channels (48B) per volume,
    // copied global->LDS with async b128 transfers (ASYNCcnt-tracked).
    const int m  = lane >> 1;
    const int c0 = (lane & 1) * 12;
    const int v  = ulist[tileBase + m];
    {
        unsigned long long ga = (unsigned long long)(uintptr_t)(gvol + (size_t)v * CCH + c0);
        unsigned long long ca = (unsigned long long)(uintptr_t)(cvol + (size_t)v * CCH + c0);
        unsigned int lh = lds_addr_of(&stg[wave][m * 48 + c0]);
        unsigned int lx = lds_addr_of(&stg[wave][m * 48 + CCH + c0]);
        async_b128(lh, ga); async_b128_off16(lh, ga); async_b128_off32(lh, ga);
        async_b128(lx, ca); async_b128_off16(lx, ca); async_b128_off32(lx, ca);
    }

    // Meanwhile stage all three gate weights (48x24 f32 row-major) -> 64x32 f16 col-major.
    const float* Wg[3] = {Wz, Wr, Wq};
#pragma unroll
    for (int g = 0; g < 3; ++g)
        for (int idx = tid; idx < 32 * 64; idx += 128) {
            int col = idx >> 6, k = idx & 63;
            float w = (col < CCH && k < 2 * CCH) ? Wg[g][k * CCH + col] : 0.0f;
            Bw[g][idx] = (_Float16)w;
        }

    wait_async0();          // this wave's staging tile is in LDS
    __syncthreads();        // weights visible block-wide

    // Convert staged f32 rows into the f16 A tile [h | x | 0-pad].
#pragma unroll
    for (int j = 0; j < 12; ++j) {
        hx[wave][m * 64 + c0 + j]       = (_Float16)stg[wave][m * 48 + c0 + j];
        hx[wave][m * 64 + CCH + c0 + j] = (_Float16)stg[wave][m * 48 + CCH + c0 + j];
    }
#pragma unroll
    for (int j = 0; j < 8; ++j)   // zero-pad K 48..63
        hx[wave][m * 64 + 48 + (lane & 1) * 8 + j] = (_Float16)0.0f;
    __syncthreads();

    // z and r gates: D = hx(16x48) * W(48x24), two K-steps, two 16-col N tiles.
    v16h a0 = load_a_frag(&hx[wave][0], lane, 0);
    v16h a1 = load_a_frag(&hx[wave][0], lane, 32);
    v8f z0 = {}, z1 = {}, r0 = {}, r1 = {};
    z0 = wmma_f16(a0, load_b_frag(Bw[0], lane, 0, 0),  z0);
    z0 = wmma_f16(a1, load_b_frag(Bw[0], lane, 0, 32), z0);
    z1 = wmma_f16(a0, load_b_frag(Bw[0], lane, 1, 0),  z1);
    z1 = wmma_f16(a1, load_b_frag(Bw[0], lane, 1, 32), z1);
    r0 = wmma_f16(a0, load_b_frag(Bw[1], lane, 0, 0),  r0);
    r0 = wmma_f16(a1, load_b_frag(Bw[1], lane, 0, 32), r0);
    r1 = wmma_f16(a0, load_b_frag(Bw[1], lane, 1, 0),  r1);
    r1 = wmma_f16(a1, load_b_frag(Bw[1], lane, 1, 32), r1);

    const int nc0 = lane & 15;                       // columns 0..15 (always valid)
    const int nc1 = 16 + (lane & 15);                // columns 16..31 (valid while < 24)
    const int nc1c = (nc1 < CCH) ? nc1 : CCH - 1;    // clamped: branch-free bias loads
    const bool hi_ok = (nc1 < CCH);
    const float bz0 = bz[nc0], br0 = br[nc0];
    const float bz1 = bz[nc1c], br1 = br[nc1c];

    float zr0[8], zr1[8];
#pragma unroll
    for (int k = 0; k < 8; ++k) {
        zr0[k] = sigmoidf_(z0[k] + bz0);
        zr1[k] = sigmoidf_(z1[k] + bz1);
        float rv0 = sigmoidf_(r0[k] + br0);
        float rv1 = sigmoidf_(r1[k] + br1);
        int mrow = k + 8 * (lane >> 4);
        // overwrite h-part of A tile with r*h (x-part stays) -> A2 = [r*h | x]
        hx[wave][mrow * 64 + nc0] = (_Float16)(rv0 * stg[wave][mrow * 48 + nc0]);
        if (hi_ok)
            hx[wave][mrow * 64 + nc1] = (_Float16)(rv1 * stg[wave][mrow * 48 + nc1]);
    }
    __syncthreads();

    // q gate on A2.
    v16h qa0 = load_a_frag(&hx[wave][0], lane, 0);
    v16h qa1 = load_a_frag(&hx[wave][0], lane, 32);
    v8f q0 = {}, q1 = {};
    q0 = wmma_f16(qa0, load_b_frag(Bw[2], lane, 0, 0),  q0);
    q0 = wmma_f16(qa1, load_b_frag(Bw[2], lane, 0, 32), q0);
    q1 = wmma_f16(qa0, load_b_frag(Bw[2], lane, 1, 0),  q1);
    q1 = wmma_f16(qa1, load_b_frag(Bw[2], lane, 1, 32), q1);

    const float bq0 = bq[nc0];
    const float bq1 = bq[nc1c];
    const int total = totalPtr[0];
#pragma unroll
    for (int k = 0; k < 8; ++k) {
        int mrow = k + 8 * (lane >> 4);
        int grow = tileBase + mrow;
        float valid = (grow < total) ? 1.0f : 0.0f;
        float hv0 = stg[wave][mrow * 48 + nc0];
        float qv0 = tanhf_(q0[k] + bq0);
        outFused[(size_t)grow * CCH + nc0] =
            ((1.0f - zr0[k]) * hv0 + zr0[k] * qv0) * valid;
        if (hi_ok) {
            float hv1 = stg[wave][mrow * 48 + nc1];
            float qv1 = tanhf_(q1[k] + bq1);
            outFused[(size_t)grow * CCH + nc1] =
                ((1.0f - zr1[k]) * hv1 + zr1[k] * qv1) * valid;
        }
    }
}

// ---------------------------------------------------------------- launcher
extern "C" void kernel_launch(void* const* d_in, const int* in_sizes, int n_in,
                              void* d_out, int out_size, void* d_ws, size_t ws_size,
                              hipStream_t stream) {
    (void)n_in; (void)out_size; (void)ws_size;
    const int*   coords     = (const int*)d_in[0];
    const float* values_in  = (const float*)d_in[1];
    const int*   gcoords    = (const int*)d_in[2];
    const float* gvals      = (const float*)d_in[3];
    const int*   relorigin  = (const int*)d_in[4];
    const float* origin     = (const float*)d_in[5];
    const float* w2ac       = (const float*)d_in[6];
    const float* Wz = (const float*)d_in[7];
    const float* bz = (const float*)d_in[8];
    const float* Wr = (const float*)d_in[9];
    const float* br = (const float*)d_in[10];
    const float* Wq = (const float*)d_in[11];
    const float* bq = (const float*)d_in[12];
    const int N = in_sizes[0] / 4;
    const int M = in_sizes[2] / 3;

    char* ws = (char*)d_ws;
    const size_t volBytes = (size_t)TOTAL * CCH * sizeof(float);
    float*         gvol   = (float*)ws;
    float*         cvol   = (float*)(ws + volBytes);
    unsigned char* mask   = (unsigned char*)(ws + 2 * volBytes);
    int*           bcnt   = (int*)(ws + 2 * volBytes + (size_t)TOTAL);
    int*           boff   = bcnt + NMASKBLK;
    int*           totalC = boff + NMASKBLK;
    int*           ulist  = totalC + 64;  // padded, keeps alignment

    float* out      = (float*)d_out;
    float* outFused = out + (size_t)TOTAL * 8;   // after coords(4) + r_coords(4)

    k_zero<<<4096, 256, 0, stream>>>((float4*)gvol, (2 * TOTAL * CCH) / 4,
                                     (int4*)ulist, TOTAL / 4);
    k_scatter_current<<<(N + 255) / 256, 256, 0, stream>>>(coords, values_in, cvol, N);
    k_scatter_global<<<(M + 255) / 256, 256, 0, stream>>>(gcoords, gvals, relorigin, gvol, M);
    k_mask<<<NMASKBLK, MASK_BLOCK, 0, stream>>>(gvol, cvol, mask, bcnt);
    k_scan<<<1, 1024, 0, stream>>>(bcnt, boff, totalC);
    k_compact<<<NMASKBLK, MASK_BLOCK, 0, stream>>>(mask, boff, ulist);
    k_coords<<<TOTAL / 256, 256, 0, stream>>>(ulist, totalC, origin, w2ac, out);
    k_gru<<<GRU_BLOCKS, 128, 0, stream>>>(gvol, cvol, ulist, totalC,
                                          Wz, bz, Wr, br, Wq, bq, outFused);
}
